// SocialTransformerEncoder_21053929685093
// MI455X (gfx1250) — matrix operand order repaired and checked by reading, not compiled, not run on hardware
//
#include <hip/hip_runtime.h>
#include <hip/hip_bf16.h>
#include <math.h>
#include <stdint.h>

// Problem constants (must match reference)
#define Bn   8192
#define Nn   64
#define Dn   128
#define Hn   8
#define HDn  16
#define Ln   3
#define Fn   512
#define WT_PITCH 136   // padded bf16 pitch for transposed weight tile (272B rows, 16B aligned)

#define USE_ASYNC_STAGE 1

typedef __attribute__((ext_vector_type(16))) __bf16 v16bf;
typedef __attribute__((ext_vector_type(8)))  __bf16 v8bf;
typedef __attribute__((ext_vector_type(8)))  float  v8f;

struct Params {
  const float* ego_context; const float* ego_pos; const float* ego_vel;
  const float* neighbor_pos; const float* neighbor_vel; const int* neighbor_mask;
  const float* np_W; const float* np_b; const float* np_ln_g; const float* np_ln_b;
  const float* ego_W; const float* ego_b; const float* ego_ln_g; const float* ego_ln_b;
  const float* n1_g; const float* n1_b; const float* nkv_g; const float* nkv_b;
  const float* n2_g; const float* n2_b;
  const float* Wq; const float* bq; const float* Wk; const float* bk;
  const float* Wv; const float* bv; const float* Wo; const float* bo;
  const float* W1; const float* b1; const float* W2; const float* b2;
  const float* out_W; const float* out_b; const float* out_ln_g; const float* out_ln_b;
  const __bf16* wkvt;   // prepped transposed bf16 K/V weights (pitch WT_PITCH), 6 tiles
  float* out;
};

__device__ __forceinline__ v16bf cat16(v8bf a, v8bf b) {
  return __builtin_shufflevector(a, b, 0,1,2,3,4,5,6,7,8,9,10,11,12,13,14,15);
}

// wave0 computes mean / rsqrt(var+eps) of v[0..n) into s_mr[0..1]; caller barriers after.
__device__ __forceinline__ void wave0_stats(const float* v, int n, float* s_mr, int tid) {
  if (tid < 32) {
    float s = 0.f, s2 = 0.f;
    for (int j = tid; j < n; j += 32) { float x = v[j]; s += x; s2 += x * x; }
    #pragma unroll
    for (int o = 16; o; o >>= 1) { s += __shfl_xor(s, o, 32); s2 += __shfl_xor(s2, o, 32); }
    if (tid == 0) {
      float m = s / (float)n;
      s_mr[0] = m;
      s_mr[1] = rsqrtf(s2 / (float)n - m * m + 1e-5f);
    }
  }
}

#if USE_ASYNC_STAGE
// Async-copy one pre-transposed bf16 weight tile (contiguous 34816 B) into LDS.
// Each issued instruction moves 16 B/lane, tracked on ASYNCcnt (caller waits + barriers).
__device__ __forceinline__ void async_stage(const __bf16* gsrc,
                                            __bf16 (*dst)[WT_PITCH], int tid) {
  const uint32_t ldsbase = (uint32_t)(uintptr_t)(&dst[0][0]);  // low 32b of generic = LDS offset
  const unsigned long long gbase = (unsigned long long)(uintptr_t)gsrc;
  const int chunks = (Dn * WT_PITCH * (int)sizeof(__bf16)) / 16;   // 2176 x b128
  for (int i = tid; i < chunks; i += 256) {
    const uint32_t go = (uint32_t)i * 16u;
    const uint32_t lo = ldsbase + go;
    asm volatile("global_load_async_to_lds_b128 %0, %1, %2"
                 :: "v"(lo), "v"(go), "s"(gbase)
                 : "memory");
  }
}
#else
// Fallback: VALU staging of a (128,128) f32 weight, transposed, into bf16 LDS tile.
__device__ __forceinline__ void stage_wT(const float* __restrict__ W,
                                         __bf16 (*s_wt)[WT_PITCH], int tid) {
  for (int idx = tid; idx < Dn * Dn; idx += 256) {
    int k = idx >> 7, n = idx & 127;
    s_wt[n][k] = (__bf16)W[idx];
  }
}
#endif

// D = A(64x128 bf16, LDS) @ W^T-tile(128x128 bf16, LDS) + bias -> s_o (bf16)
// 8 waves cover the 4x8 grid of 16x16 tiles; K=128 in 4 steps of 32 (bf16 WMMA).
__device__ __forceinline__ void wmma_proj(const __bf16 (*s_a)[Dn],
                                          const __bf16 (*s_w)[WT_PITCH],
                                          const float* __restrict__ bias,
                                          __bf16 (*s_o)[Dn], int tid) {
  const int lane   = tid & 31;
  const int wv     = tid >> 5;
  const int mt     = wv & 3;            // M tile 0..3
  const int ntBase = (wv >> 2) * 4;     // N tiles [ntBase, ntBase+3]
  const int mrow   = mt * 16 + (lane & 15);
  const int abase  = (lane >> 4) * 8;   // A-layout k base per lane half
  const int ncol   = lane & 15;
  const int bko    = (lane >> 4) * 8;   // B-layout k base per lane half

  v8f acc[4] = {};
  #pragma unroll
  for (int ks = 0; ks < 4; ++ks) {
    const int k0 = ks * 32 + abase;
    v8bf alo = *(const v8bf*)&s_a[mrow][k0];
    v8bf ahi = *(const v8bf*)&s_a[mrow][k0 + 16];
    v16bf A = cat16(alo, ahi);
    #pragma unroll
    for (int j = 0; j < 4; ++j) {
      const int n  = (ntBase + j) * 16 + ncol;
      const int kb = ks * 32 + bko;
      v8bf blo = *(const v8bf*)&s_w[n][kb];
      v8bf bhi = *(const v8bf*)&s_w[n][kb + 16];
      v16bf Bm = cat16(blo, bhi);
      acc[j] = __builtin_amdgcn_wmma_f32_16x16x32_bf16(
          false, A, false, Bm, (short)0, acc[j], false, false);
    }
  }
  #pragma unroll
  for (int j = 0; j < 4; ++j) {
    const int n = (ntBase + j) * 16 + ncol;
    const float bsv = bias[n];
    #pragma unroll
    for (int v = 0; v < 8; ++v) {
      const int m = mt * 16 + (lane >> 4) * 8 + v;   // C/D VGPR layout
      s_o[m][n] = (__bf16)(acc[j][v] + bsv);
    }
  }
}

// Prep: write WkT/WvT (bf16, pitch WT_PITCH, contiguous per tile) into workspace.
// Tile t = li*2 + {0:K, 1:V}; ws tile layout matches the LDS tile byte-for-byte.
__global__ __launch_bounds__(256)
void prep_weights_kernel(const float* __restrict__ Wk, const float* __restrict__ Wv,
                         __bf16* __restrict__ ws) {
  const int idx = blockIdx.x * 256 + threadIdx.x;
  if (idx >= Ln * Dn * Dn) return;
  const int li = idx / (Dn * Dn);
  const int r  = idx - li * Dn * Dn;
  const int k  = r >> 7, n = r & 127;
  ws[((size_t)(li * 2 + 0) * Dn + n) * WT_PITCH + k] = (__bf16)Wk[idx];
  ws[((size_t)(li * 2 + 1) * Dn + n) * WT_PITCH + k] = (__bf16)Wv[idx];
}

__global__ __launch_bounds__(256, 2)
void social_encoder_kernel(Params p) {
  const int tid  = threadIdx.x;
  const int b    = blockIdx.x;
  const int lane = tid & 31;
  const int wv   = tid >> 5;

  __shared__ __bf16 s_nb[Nn][Dn];            // 16 KB  post-ReLU neighbor features
  __shared__ __bf16 s_kv[Nn][Dn];            // 16 KB  per-layer LN'd A operand
  __shared__ __bf16 s_wt[2][Dn][WT_PITCH];   // 69.6KB double-buffered WkT/WvT tiles
  __shared__ __bf16 s_K[Nn][Dn];             // 16 KB
  __shared__ __bf16 s_V[Nn][Dn];             // 16 KB
  __shared__ float  s_rel[Nn][8];
  __shared__ int    s_valid[Nn];
  __shared__ float  s_nbm[Nn], s_nbr[Nn];    // precomputed per-row LN stats of nb
  __shared__ float  s_ego[Dn];
  __shared__ float  s_x[Dn], s_q[Dn], s_Q[Dn], s_ctx[Dn];
  __shared__ float  s_ffn[Fn];
  __shared__ float  s_attn[Hn][Nn];
  __shared__ float  s_mr[2];

  // ---- Phase A: load ego row, relative features, validity mask ----
  if (tid < Dn) s_ego[tid] = p.ego_context[(long)b * Dn + tid];
  if (tid < Nn) {
    const int n = tid;
    const float ex  = p.ego_pos[b * 2 + 0], ey  = p.ego_pos[b * 2 + 1];
    const float evx = p.ego_vel[b * 2 + 0], evy = p.ego_vel[b * 2 + 1];
    const long base = ((long)b * Nn + n) * 2;
    const float rx  = p.neighbor_pos[base + 0] - ex;
    const float ry  = p.neighbor_pos[base + 1] - ey;
    const float rvx = p.neighbor_vel[base + 0] - evx;
    const float rvy = p.neighbor_vel[base + 1] - evy;
    const float nrm  = sqrtf(rx * rx + ry * ry);
    const float dist = fmaxf(nrm, 1e-6f);
    const float bear = atan2f(ry + 1e-8f, rx + 1e-8f);
    s_rel[n][0] = rx;  s_rel[n][1] = ry;
    s_rel[n][2] = rvx; s_rel[n][3] = rvy;
    s_rel[n][4] = dist; s_rel[n][5] = bear;
    s_valid[n] = (p.neighbor_mask[(long)b * Nn + n] != 0) && (nrm < 3.0f);
  }
  __syncthreads();

  // ---- Phase B: valid fixup, nb pre-activation, ego projection ----
  if (tid == 0) {
    int any = 0;
    for (int n = 0; n < Nn; ++n) any |= s_valid[n];
    if (!any) s_valid[0] = 1;
  }
  {
    const int n = tid >> 2, fb = (tid & 3) * 32;   // 4 threads per neighbor row
    const float r0 = s_rel[n][0], r1 = s_rel[n][1], r2 = s_rel[n][2];
    const float r3 = s_rel[n][3], r4 = s_rel[n][4], r5 = s_rel[n][5];
    for (int j = 0; j < 32; ++j) {
      const int f = fb + j;
      float a = r0 * p.np_W[0 * Dn + f] + r1 * p.np_W[1 * Dn + f]
              + r2 * p.np_W[2 * Dn + f] + r3 * p.np_W[3 * Dn + f]
              + r4 * p.np_W[4 * Dn + f] + r5 * p.np_W[5 * Dn + f] + p.np_b[f];
      s_nb[n][f] = (__bf16)a;
    }
  }
  if (tid < Dn) {
    float a = p.ego_b[tid];
    #pragma unroll 8
    for (int k = 0; k < Dn; ++k) a += s_ego[k] * p.ego_W[k * Dn + tid];
    s_x[tid] = a;
  }
  __syncthreads();

  // ---- Phase C: neighbor-proj LN+ReLU (+ per-row stats of result), ego LN ----
  if (tid < Nn) {
    const int n = tid;
    float s = 0.f, s2 = 0.f;
    for (int f = 0; f < Dn; ++f) { float v = (float)s_nb[n][f]; s += v; s2 += v * v; }
    const float m = s * (1.f / Dn);
    const float rstd = rsqrtf(s2 * (1.f / Dn) - m * m + 1e-5f);
    float ps = 0.f, ps2 = 0.f;
    for (int f = 0; f < Dn; ++f) {
      float v = ((float)s_nb[n][f] - m) * rstd * p.np_ln_g[f] + p.np_ln_b[f];
      v = fmaxf(v, 0.f);
      s_nb[n][f] = (__bf16)v;
      ps += v; ps2 += v * v;
    }
    const float m2 = ps * (1.f / Dn);
    s_nbm[n] = m2;
    s_nbr[n] = rsqrtf(ps2 * (1.f / Dn) - m2 * m2 + 1e-5f);
  }
  wave0_stats(s_x, Dn, s_mr, tid);
  __syncthreads();
  if (tid < Dn)
    s_x[tid] = (s_x[tid] - s_mr[0]) * s_mr[1] * p.ego_ln_g[tid] + p.ego_ln_b[tid];
  __syncthreads();

  const float scale = 0.25f;   // 1/sqrt(HD=16)

  for (int li = 0; li < Ln; ++li) {
    const float* Wq_l = p.Wq + (long)li * Dn * Dn; const float* bq_l = p.bq + li * Dn;
    const float* bk_l = p.bk + li * Dn;
    const float* bv_l = p.bv + li * Dn;
    const float* Wo_l = p.Wo + (long)li * Dn * Dn; const float* bo_l = p.bo + li * Dn;
    const float* W1_l = p.W1 + (long)li * Dn * Fn; const float* b1_l = p.b1 + li * Fn;
    const float* W2_l = p.W2 + (long)li * Fn * Dn; const float* b2_l = p.b2 + li * Dn;

    // Kick off async weight-tile copies (WkT -> buf0, WvT -> buf1), overlapped
    // with the kv = LN(nb) fill below.
#if USE_ASYNC_STAGE
    async_stage(p.wkvt + (size_t)(li * 2 + 0) * Dn * WT_PITCH, s_wt[0], tid);
    async_stage(p.wkvt + (size_t)(li * 2 + 1) * Dn * WT_PITCH, s_wt[1], tid);
#else
    stage_wT(p.Wk + (long)li * Dn * Dn, s_wt[0], tid);
    stage_wT(p.Wv + (long)li * Dn * Dn, s_wt[1], tid);
#endif

    // kv = LN(nb) with precomputed stats -> bf16 A operand
    {
      const int n = tid >> 2, fb = (tid & 3) * 32;
      const float m = s_nbm[n], r = s_nbr[n];
      for (int j = 0; j < 32; ++j) {
        const int f = fb + j;
        float v = ((float)s_nb[n][f] - m) * r * p.nkv_g[li * Dn + f] + p.nkv_b[li * Dn + f];
        s_kv[n][f] = (__bf16)v;
      }
    }
#if USE_ASYNC_STAGE
    asm volatile("s_wait_asynccnt 0x0" ::: "memory");
#endif
    __syncthreads();

    // K = kv @ Wk + bk, V = kv @ Wv + bv  (WMMA, back to back)
    wmma_proj(s_kv, s_wt[0], bk_l, s_K, tid);
    wmma_proj(s_kv, s_wt[1], bv_l, s_V, tid);
    __syncthreads();

    // q = LN(x; n1)
    wave0_stats(s_x, Dn, s_mr, tid);
    __syncthreads();
    if (tid < Dn)
      s_q[tid] = (s_x[tid] - s_mr[0]) * s_mr[1] * p.n1_g[li * Dn + tid] + p.n1_b[li * Dn + tid];
    __syncthreads();

    // Q = q @ Wq + bq  (gemv, weights from L2)
    if (tid < Dn) {
      float a = bq_l[tid];
      #pragma unroll 8
      for (int k = 0; k < Dn; ++k) a += s_q[k] * Wq_l[k * Dn + tid];
      s_Q[tid] = a;
    }
    __syncthreads();

    // scores (masked)
    for (int pI = tid; pI < Hn * Nn; pI += 256) {
      const int h = pI >> 6, n = pI & 63;
      float s = 0.f;
      #pragma unroll
      for (int d = 0; d < HDn; ++d) s += s_Q[h * HDn + d] * (float)s_K[n][h * HDn + d];
      s *= scale;
      if (!s_valid[n]) s = -1e9f;
      s_attn[h][n] = s;
    }
    __syncthreads();

    // softmax: one wave per head (64 values -> 2 per lane)
    {
      float v0 = s_attn[wv][lane], v1 = s_attn[wv][lane + 32];
      float mx = fmaxf(v0, v1);
      #pragma unroll
      for (int o = 16; o; o >>= 1) mx = fmaxf(mx, __shfl_xor(mx, o, 32));
      float e0 = expf(v0 - mx), e1 = expf(v1 - mx);
      float ssum = e0 + e1;
      #pragma unroll
      for (int o = 16; o; o >>= 1) ssum += __shfl_xor(ssum, o, 32);
      const float inv = 1.f / ssum;
      s_attn[wv][lane]      = e0 * inv;
      s_attn[wv][lane + 32] = e1 * inv;
    }
    __syncthreads();

    // ctx = attn @ V
    if (tid < Dn) {
      const int h = tid >> 4, d = tid & 15;
      float c = 0.f;
      for (int n = 0; n < Nn; ++n) c += s_attn[h][n] * (float)s_V[n][h * HDn + d];
      s_ctx[tid] = c;
    }
    __syncthreads();

    // x += ctx @ Wo + bo
    if (tid < Dn) {
      float a = bo_l[tid];
      #pragma unroll 8
      for (int k = 0; k < Dn; ++k) a += s_ctx[k] * Wo_l[k * Dn + tid];
      s_x[tid] += a;
    }
    __syncthreads();

    // y = LN(x; n2)
    wave0_stats(s_x, Dn, s_mr, tid);
    __syncthreads();
    if (tid < Dn)
      s_q[tid] = (s_x[tid] - s_mr[0]) * s_mr[1] * p.n2_g[li * Dn + tid] + p.n2_b[li * Dn + tid];
    __syncthreads();

    // FFN1: gelu(y @ W1 + b1)  (exact gelu)
    for (int f = tid; f < Fn; f += 256) {
      float a = b1_l[f];
      #pragma unroll 8
      for (int k = 0; k < Dn; ++k) a += s_q[k] * W1_l[k * Fn + f];
      s_ffn[f] = 0.5f * a * (1.f + erff(a * 0.70710678118f));
    }
    __syncthreads();

    // FFN2: x += h @ W2 + b2
    if (tid < Dn) {
      float a = b2_l[tid];
      #pragma unroll 8
      for (int k = 0; k < Fn; ++k) a += s_ffn[k] * W2_l[k * Dn + tid];
      s_x[tid] += a;
    }
    __syncthreads();
  }

  // ---- Output: LN(x @ out_W + out_b) ----
  if (tid < Dn) {
    float a = p.out_b[tid];
    #pragma unroll 8
    for (int k = 0; k < Dn; ++k) a += s_x[k] * p.out_W[k * Dn + tid];
    s_q[tid] = a;
  }
  __syncthreads();
  wave0_stats(s_q, Dn, s_mr, tid);
  __syncthreads();
  if (tid < Dn)
    p.out[(long)b * Dn + tid] =
        (s_q[tid] - s_mr[0]) * s_mr[1] * p.out_ln_g[tid] + p.out_ln_b[tid];
}

extern "C" void kernel_launch(void* const* d_in, const int* in_sizes, int n_in,
                              void* d_out, int out_size, void* d_ws, size_t ws_size,
                              hipStream_t stream) {
  (void)in_sizes; (void)n_in; (void)out_size; (void)ws_size;
  Params p;
  p.ego_context  = (const float*)d_in[0];
  p.ego_pos      = (const float*)d_in[1];
  p.ego_vel      = (const float*)d_in[2];
  p.neighbor_pos = (const float*)d_in[3];
  p.neighbor_vel = (const float*)d_in[4];
  p.neighbor_mask= (const int*)  d_in[5];
  p.np_W   = (const float*)d_in[6];  p.np_b   = (const float*)d_in[7];
  p.np_ln_g= (const float*)d_in[8];  p.np_ln_b= (const float*)d_in[9];
  p.ego_W  = (const float*)d_in[10]; p.ego_b  = (const float*)d_in[11];
  p.ego_ln_g=(const float*)d_in[12]; p.ego_ln_b=(const float*)d_in[13];
  p.n1_g   = (const float*)d_in[14]; p.n1_b   = (const float*)d_in[15];
  p.nkv_g  = (const float*)d_in[16]; p.nkv_b  = (const float*)d_in[17];
  p.n2_g   = (const float*)d_in[18]; p.n2_b   = (const float*)d_in[19];
  p.Wq = (const float*)d_in[20]; p.bq = (const float*)d_in[21];
  p.Wk = (const float*)d_in[22]; p.bk = (const float*)d_in[23];
  p.Wv = (const float*)d_in[24]; p.bv = (const float*)d_in[25];
  p.Wo = (const float*)d_in[26]; p.bo = (const float*)d_in[27];
  p.W1 = (const float*)d_in[28]; p.b1 = (const float*)d_in[29];
  p.W2 = (const float*)d_in[30]; p.b2 = (const float*)d_in[31];
  p.out_W  = (const float*)d_in[32]; p.out_b  = (const float*)d_in[33];
  p.out_ln_g=(const float*)d_in[34]; p.out_ln_b=(const float*)d_in[35];
  p.wkvt = (const __bf16*)d_ws;     // needs 3*2*128*136*2 = 208,896 B of workspace
  p.out  = (float*)d_out;

  // Pass 1: transpose + bf16-convert K/V weights into workspace (L2-resident).
  prep_weights_kernel<<<dim3((Ln * Dn * Dn + 255) / 256), dim3(256), 0, stream>>>(
      (const float*)d_in[22], (const float*)d_in[24], (__bf16*)d_ws);

  // Pass 2: fused encoder, one workgroup per batch element.
  social_encoder_kernel<<<dim3(Bn), dim3(256), 0, stream>>>(p);
}
